// AllAtomAtomConvolution_8461085573572
// MI455X (gfx1250) — matrix operand order repaired and checked by reading, not compiled, not run on hardware
//
#include <hip/hip_runtime.h>
#include <hip/hip_bf16.h>

typedef __attribute__((ext_vector_type(16))) _Float16 v16h;
typedef __attribute__((ext_vector_type(8)))  _Float16 v8h;
typedef __attribute__((ext_vector_type(4)))  _Float16 v4h;
typedef __attribute__((ext_vector_type(8)))  float    v8f;

#define PI_F  3.14159265358979f
#define CUT_F 5.0f
// RBF: centers i*(5/15), coeff = -0.5/(5/15)^2 = -4.5

__device__ __forceinline__ float silu_f(float x){ return x * (1.0f/(1.0f+__expf(-x))); }
__device__ __forceinline__ float sigm_f(float x){ return 1.0f/(1.0f+__expf(-x)); }

__device__ __forceinline__ v4h cvt4(float4 v){
  v4h o; o[0]=(_Float16)v.x; o[1]=(_Float16)v.y; o[2]=(_Float16)v.z; o[3]=(_Float16)v.w;
  return o;
}

// --- WMMA fragment helpers (wave32, 16x16x32 f16) -------------------------
// A: 16xK tile, row-major f16 in LDS, leading dim ldk (halfs).
// Layout per ISA 7.12.2: lane L (row = L&15, hi = L>>4):
//   elems 0..7  -> K = kt*32 + hi*8 + j ; elems 8..15 -> +16
__device__ __forceinline__ v16h a_frag_lds(const _Float16* tile, int ldk, int kt, int lane){
  int row = lane & 15, hi = (lane >> 4) & 1;
  const _Float16* p = tile + row*ldk + kt*32 + hi*8;
  v8h lo = *(const v8h*)p;
  v8h hh = *(const v8h*)(p + 16);
  return __builtin_shufflevector(lo, hh, 0,1,2,3,4,5,6,7,8,9,10,11,12,13,14,15);
}

// B: pre-packed fragment-major: frag (kt,nt) = 32 lanes x 16 halfs contiguous.
__device__ __forceinline__ v16h b_frag_g(const _Float16* packed, int KT, int kt, int nt, int lane){
  return *(const v16h*)(packed + ((size_t)(nt*KT + kt)*32 + lane)*16);
}

__device__ __forceinline__ v8f wmma16(v16h a, v16h b, v8f c){
  return __builtin_amdgcn_wmma_f32_16x16x32_f16(false, a, false, b, (short)0, c, false, false);
}

// --- utility kernels ------------------------------------------------------
__global__ void zero_kernel(float* p, long n){
  long i = (long)blockIdx.x*blockDim.x + threadIdx.x;
  if (i < n) p[i] = 0.0f;
}

// Pack f32 weight W[Korig][Nf] into WMMA-B fragment-major f16, K padded to KT*32.
// elem j of lane l in frag (kt,nt):  K = kt*32 + (l>=16?8:0) + (j&7) + (j>=8?16:0),  N = nt*16 + (l&15)
__global__ void pack_kernel(const float* __restrict__ W, _Float16* __restrict__ P,
                            int Korig, int Nf, int KT){
  long t = (long)blockIdx.x*blockDim.x + threadIdx.x;
  long total = (long)KT * (Nf/16) * 512;
  if (t >= total) return;
  int j    = (int)(t & 15);
  int lane = (int)((t >> 4) & 31);
  long tile = t >> 9;
  int kt = (int)(tile % KT);
  int nt = (int)(tile / KT);
  int K = kt*32 + ((lane >= 16) ? 8 : 0) + (j & 7) + ((j >= 8) ? 16 : 0);
  int n = nt*16 + (lane & 15);
  float v = (K < Korig) ? W[(long)K*Nf + n] : 0.0f;
  P[t] = (_Float16)v;
}

// e_gate = MLP3(e_feat): [100,64] -> 256 -> 256 -> 128, tiny -> VALU
__global__ __launch_bounds__(256) void egate_kernel(
    const float* __restrict__ e_feat,
    const float* __restrict__ ew1, const float* __restrict__ eb1,
    const float* __restrict__ ew2, const float* __restrict__ eb2,
    const float* __restrict__ ew3, const float* __restrict__ eb3,
    float* __restrict__ egate){
  __shared__ float x1[256];
  __shared__ float x2[256];
  int t = threadIdx.x;
  for (int r = 0; r < 100; ++r){
    float s = eb1[t];
    for (int k = 0; k < 64; ++k) s += e_feat[r*64+k] * ew1[k*256+t];
    x1[t] = silu_f(s);
    __syncthreads();
    float s2 = eb2[t];
    for (int k = 0; k < 256; ++k) s2 += x1[k] * ew2[k*256+t];
    x2[t] = silu_f(s2);
    __syncthreads();
    if (t < 128){
      float s3 = eb3[t];
      for (int k = 0; k < 256; ++k) s3 += x2[k] * ew3[k*128+t];
      egate[r*128+t] = s3;
    }
    __syncthreads();
  }
}

// --- edge kernel: 1 wave/block, 32-edge tile (2 M-subtiles, B reused 2x) --
__global__ __launch_bounds__(32) void edge_kernel(
    const float* __restrict__ h, const int* __restrict__ z,
    const unsigned char* __restrict__ mask,
    const int* __restrict__ att_src, const int* __restrict__ att_dst,
    const float* __restrict__ att_dist, const float* __restrict__ ze,
    const _Float16* __restrict__ mw1p, const float* __restrict__ mb1,
    const _Float16* __restrict__ mw2p, const float* __restrict__ mb2,
    const _Float16* __restrict__ mw3p, const float* __restrict__ mb3,
    const _Float16* __restrict__ gw1p, const float* __restrict__ gb1,
    const float* __restrict__ gw2,  const float* __restrict__ gb2,
    float* __restrict__ accum){
  __shared__ __align__(16) unsigned char smemA[32*288*2];   // f16 [32][288]
  __shared__ __align__(16) unsigned char smemB[32*128*4];   // f16 [32][256] or f32 [32][128]
  __shared__ float sCoef[32];

  const int lane = threadIdx.x & 31;
  const int e0   = blockIdx.x * 32;

  _Float16* A  = (_Float16*)smemA;
  _Float16* Bh = (_Float16*)smemB;
  float*    Bf = (float*)smemB;

  const int half = lane >> 4;
  const int nloc = lane & 15;

  // one row per lane
  const int   src_r  = att_src[e0 + lane];
  const int   dst_r  = att_dst[e0 + lane];
  const float d_r    = att_dist[e0 + lane];
  const float self_r = (src_r == dst_r) ? 1.0f : 0.0f;
  const int   zrow   = z[dst_r];

  __builtin_prefetch(att_src + e0 + 32, 0, 0);
  __builtin_prefetch(att_dst + e0 + 32, 0, 0);
  __builtin_prefetch(att_dist + e0 + 32, 0, 0);

  // ---- build msg_in [32][192]: h_dst(128) | ze[z_dst](32) | is_self | rbf(16) | pad
  {
    _Float16* Ar = A + lane*192;
    const float4* hp = (const float4*)(h + (size_t)dst_r*128);
    #pragma unroll 8
    for (int c4 = 0; c4 < 32; ++c4) *(v4h*)(Ar + c4*4) = cvt4(hp[c4]);
    const float4* zp = (const float4*)(ze + (size_t)zrow*32);
    #pragma unroll
    for (int c4 = 0; c4 < 8; ++c4) *(v4h*)(Ar + 128 + c4*4) = cvt4(zp[c4]);
    Ar[160] = (_Float16)self_r;
    #pragma unroll
    for (int i = 0; i < 16; ++i){
      float ctr = (float)i*(CUT_F/15.0f); float tt = d_r - ctr;
      Ar[161+i] = (_Float16)__expf(-4.5f*tt*tt);
    }
    #pragma unroll
    for (int i = 177; i < 192; ++i) Ar[i] = (_Float16)0.0f;
  }
  __syncthreads();

  // ---- msg layer1: 192 -> 256 (silu)
  v16h a0[2][6];
  #pragma unroll
  for (int s = 0; s < 2; ++s)
    #pragma unroll
    for (int kt = 0; kt < 6; ++kt) a0[s][kt] = a_frag_lds(A + s*16*192, 192, kt, lane);
  __syncthreads();
  __builtin_prefetch(mw2p, 0, 1);
  for (int nt = 0; nt < 16; ++nt){
    v8f c0 = {}, c1 = {};
    #pragma unroll
    for (int kt = 0; kt < 6; ++kt){
      v16h b = b_frag_g(mw1p, 6, kt, nt, lane);
      c0 = wmma16(a0[0][kt], b, c0);
      c1 = wmma16(a0[1][kt], b, c1);
    }
    float bb = mb1[nt*16 + nloc];
    #pragma unroll
    for (int r = 0; r < 8; ++r){
      Bh[(half*8 + r)*256      + nt*16 + nloc] = (_Float16)silu_f(c0[r] + bb);
      Bh[(16 + half*8 + r)*256 + nt*16 + nloc] = (_Float16)silu_f(c1[r] + bb);
    }
  }
  __syncthreads();

  // ---- msg layer2: 256 -> 256 (silu), output into A (reused, ld 256)
  v16h a1[2][8];
  #pragma unroll
  for (int s = 0; s < 2; ++s)
    #pragma unroll
    for (int kt = 0; kt < 8; ++kt) a1[s][kt] = a_frag_lds(Bh + s*16*256, 256, kt, lane);
  __syncthreads();
  __builtin_prefetch(mw3p, 0, 1);
  for (int nt = 0; nt < 16; ++nt){
    v8f c0 = {}, c1 = {};
    #pragma unroll
    for (int kt = 0; kt < 8; ++kt){
      v16h b = b_frag_g(mw2p, 8, kt, nt, lane);
      c0 = wmma16(a1[0][kt], b, c0);
      c1 = wmma16(a1[1][kt], b, c1);
    }
    float bb = mb2[nt*16 + nloc];
    #pragma unroll
    for (int r = 0; r < 8; ++r){
      A[(half*8 + r)*256      + nt*16 + nloc] = (_Float16)silu_f(c0[r] + bb);
      A[(16 + half*8 + r)*256 + nt*16 + nloc] = (_Float16)silu_f(c1[r] + bb);
    }
  }
  __syncthreads();

  // ---- msg layer3: 256 -> 128 (linear), f32 msg into Bf [32][128]
  v16h a2[2][8];
  #pragma unroll
  for (int s = 0; s < 2; ++s)
    #pragma unroll
    for (int kt = 0; kt < 8; ++kt) a2[s][kt] = a_frag_lds(A + s*16*256, 256, kt, lane);
  __syncthreads();
  __builtin_prefetch(gw1p, 0, 1);
  for (int nt = 0; nt < 8; ++nt){
    v8f c0 = {}, c1 = {};
    #pragma unroll
    for (int kt = 0; kt < 8; ++kt){
      v16h b = b_frag_g(mw3p, 8, kt, nt, lane);
      c0 = wmma16(a2[0][kt], b, c0);
      c1 = wmma16(a2[1][kt], b, c1);
    }
    float bb = mb3[nt*16 + nloc];
    #pragma unroll
    for (int r = 0; r < 8; ++r){
      Bf[(half*8 + r)*128      + nt*16 + nloc] = c0[r] + bb;
      Bf[(16 + half*8 + r)*128 + nt*16 + nloc] = c1[r] + bb;
    }
  }
  __syncthreads();

  // ---- build gate_in [32][288]: h_src(128)|h_dst(128)|rbf(16)|is_self|pad
  {
    _Float16* Gr = A + lane*288;
    const float4* hs = (const float4*)(h + (size_t)src_r*128);
    const float4* hd = (const float4*)(h + (size_t)dst_r*128);
    #pragma unroll 8
    for (int c4 = 0; c4 < 32; ++c4) *(v4h*)(Gr + c4*4)       = cvt4(hs[c4]);
    #pragma unroll 8
    for (int c4 = 0; c4 < 32; ++c4) *(v4h*)(Gr + 128 + c4*4) = cvt4(hd[c4]);
    #pragma unroll
    for (int i = 0; i < 16; ++i){
      float ctr = (float)i*(CUT_F/15.0f); float tt = d_r - ctr;
      Gr[256+i] = (_Float16)__expf(-4.5f*tt*tt);
    }
    Gr[272] = (_Float16)self_r;
    #pragma unroll
    for (int i = 273; i < 288; ++i) Gr[i] = (_Float16)0.0f;
  }
  __syncthreads();

  // ---- gate layer1 (288 -> 256, silu) fused with layer2 dot (256 -> 1)
  v16h g[2][9];
  #pragma unroll
  for (int s = 0; s < 2; ++s)
    #pragma unroll
    for (int kt = 0; kt < 9; ++kt) g[s][kt] = a_frag_lds(A + s*16*288, 288, kt, lane);
  float gp[2][8];
  #pragma unroll
  for (int r = 0; r < 8; ++r){ gp[0][r] = 0.0f; gp[1][r] = 0.0f; }
  for (int nt = 0; nt < 16; ++nt){
    v8f c0 = {}, c1 = {};
    #pragma unroll
    for (int kt = 0; kt < 9; ++kt){
      v16h b = b_frag_g(gw1p, 9, kt, nt, lane);
      c0 = wmma16(g[0][kt], b, c0);
      c1 = wmma16(g[1][kt], b, c1);
    }
    float bb = gb1[nt*16 + nloc];
    float w2 = gw2[nt*16 + nloc];
    #pragma unroll
    for (int r = 0; r < 8; ++r){
      gp[0][r] += silu_f(c0[r] + bb) * w2;
      gp[1][r] += silu_f(c1[r] + bb) * w2;
    }
  }
  // reduce the 16 columns held across lanes (xor within each 16-lane half)
  #pragma unroll
  for (int off = 1; off < 16; off <<= 1){
    #pragma unroll
    for (int r = 0; r < 8; ++r){
      gp[0][r] += __shfl_xor(gp[0][r], off, 32);
      gp[1][r] += __shfl_xor(gp[1][r], off, 32);
    }
  }
  if (nloc == 0){
    float b2 = gb2[0];
    #pragma unroll
    for (int r = 0; r < 8; ++r){
      sCoef[half*8 + r]      = sigm_f(gp[0][r] + b2);
      sCoef[16 + half*8 + r] = sigm_f(gp[1][r] + b2);
    }
  }
  __syncthreads();
  {
    float cut = (d_r < CUT_F) ? 0.5f*(__cosf(PI_F*d_r/CUT_F) + 1.0f) : 0.0f;
    float act = (mask[src_r] && mask[dst_r]) ? 1.0f : 0.0f;
    sCoef[lane] *= cut * act;
  }
  __syncthreads();

  // ---- scatter-add msg*coef into accum[att_src]
  for (int r = 0; r < 32; ++r){
    int   s  = __shfl(src_r, r, 32);
    float cf = sCoef[r];
    #pragma unroll
    for (int c = lane; c < 128; c += 32)
      atomicAdd(&accum[(long)s*128 + c], Bf[r*128 + c] * cf);
  }
}

// --- output kernel: 2 waves/block, 32 rows/wave; R = f*100 + e ------------
__global__ __launch_bounds__(64) void out_kernel(
    const float* __restrict__ accum, const float* __restrict__ egate,
    const _Float16* __restrict__ ow1p, const float* __restrict__ ob1,
    const _Float16* __restrict__ ow2p, const float* __restrict__ ob2,
    float* __restrict__ out){
  __shared__ __align__(16) _Float16 sA[2][32*128];
  __shared__ __align__(16) _Float16 sB[2][32*256];

  const int wave = threadIdx.x >> 5;
  const int lane = threadIdx.x & 31;
  _Float16* A  = sA[wave];
  _Float16* Bh = sB[wave];

  const long R0  = ((long)blockIdx.x*2 + wave)*32;
  const int half = lane >> 4;
  const int nloc = lane & 15;

  // build x = accum[f] * egate[e], one row per lane
  {
    long R = R0 + lane;
    int f = (int)(R / 100);
    int e = (int)(R % 100);
    const float4* ap = (const float4*)(accum + (size_t)f*128);
    const float4* gp = (const float4*)(egate + (size_t)e*128);
    _Float16* Ar = A + lane*128;
    #pragma unroll 8
    for (int c4 = 0; c4 < 32; ++c4){
      float4 u = ap[c4], w = gp[c4];
      v4h o; o[0]=(_Float16)(u.x*w.x); o[1]=(_Float16)(u.y*w.y);
             o[2]=(_Float16)(u.z*w.z); o[3]=(_Float16)(u.w*w.w);
      *(v4h*)(Ar + c4*4) = o;
    }
  }
  __syncthreads();

  v16h a[2][4];
  #pragma unroll
  for (int s = 0; s < 2; ++s)
    #pragma unroll
    for (int kt = 0; kt < 4; ++kt) a[s][kt] = a_frag_lds(A + s*16*128, 128, kt, lane);
  for (int nt = 0; nt < 16; ++nt){
    v8f c0 = {}, c1 = {};
    #pragma unroll
    for (int kt = 0; kt < 4; ++kt){
      v16h b = b_frag_g(ow1p, 4, kt, nt, lane);
      c0 = wmma16(a[0][kt], b, c0);
      c1 = wmma16(a[1][kt], b, c1);
    }
    float bb = ob1[nt*16 + nloc];
    #pragma unroll
    for (int r = 0; r < 8; ++r){
      Bh[(half*8 + r)*256      + nt*16 + nloc] = (_Float16)silu_f(c0[r] + bb);
      Bh[(16 + half*8 + r)*256 + nt*16 + nloc] = (_Float16)silu_f(c1[r] + bb);
    }
  }
  __syncthreads();

  v16h a2[2][8];
  #pragma unroll
  for (int s = 0; s < 2; ++s)
    #pragma unroll
    for (int kt = 0; kt < 8; ++kt) a2[s][kt] = a_frag_lds(Bh + s*16*256, 256, kt, lane);
  for (int nt = 0; nt < 8; ++nt){
    v8f c0 = {}, c1 = {};
    #pragma unroll
    for (int kt = 0; kt < 8; ++kt){
      v16h b = b_frag_g(ow2p, 8, kt, nt, lane);
      c0 = wmma16(a2[0][kt], b, c0);
      c1 = wmma16(a2[1][kt], b, c1);
    }
    float bb = ob2[nt*16 + nloc];
    #pragma unroll
    for (int r = 0; r < 8; ++r){
      out[(R0 + half*8 + r)*128      + nt*16 + nloc] = c0[r] + bb;
      out[(R0 + 16 + half*8 + r)*128 + nt*16 + nloc] = c1[r] + bb;
    }
  }
}

// --- host-side launcher ---------------------------------------------------
extern "C" void kernel_launch(void* const* d_in, const int* in_sizes, int n_in,
                              void* d_out, int out_size, void* d_ws, size_t ws_size,
                              hipStream_t stream){
  (void)in_sizes; (void)n_in; (void)out_size; (void)ws_size;
  const float*         h        = (const float*)d_in[0];
  const int*           z        = (const int*)d_in[1];
  const unsigned char* mask     = (const unsigned char*)d_in[2];
  const float*         e_feat   = (const float*)d_in[3];
  const int*           att_src  = (const int*)d_in[4];
  const int*           att_dst  = (const int*)d_in[5];
  const float*         att_dist = (const float*)d_in[6];
  const float*         ze       = (const float*)d_in[7];
  const float* mw1 = (const float*)d_in[8];  const float* mb1 = (const float*)d_in[9];
  const float* mw2 = (const float*)d_in[10]; const float* mb2 = (const float*)d_in[11];
  const float* mw3 = (const float*)d_in[12]; const float* mb3 = (const float*)d_in[13];
  const float* gw1 = (const float*)d_in[14]; const float* gb1 = (const float*)d_in[15];
  const float* gw2 = (const float*)d_in[16]; const float* gb2 = (const float*)d_in[17];
  const float* ew1 = (const float*)d_in[18]; const float* eb1 = (const float*)d_in[19];
  const float* ew2 = (const float*)d_in[20]; const float* eb2 = (const float*)d_in[21];
  const float* ew3 = (const float*)d_in[22]; const float* eb3 = (const float*)d_in[23];
  const float* ow1 = (const float*)d_in[24]; const float* ob1 = (const float*)d_in[25];
  const float* ow2 = (const float*)d_in[26]; const float* ob2 = (const float*)d_in[27];

  char* ws = (char*)d_ws;
  float*    accum = (float*)(ws + 0);          // 2048*128 f32 = 1,048,576 B
  float*    egate = (float*)(ws + 1048576);    // 100*128 f32  =    51,200 B
  _Float16* mw1p  = (_Float16*)(ws + 1099776); // 192*256 f16  =    98,304 B
  _Float16* mw2p  = (_Float16*)(ws + 1198080); // 256*256 f16  =   131,072 B
  _Float16* mw3p  = (_Float16*)(ws + 1329152); // 256*128 f16  =    65,536 B
  _Float16* gw1p  = (_Float16*)(ws + 1394688); // 288*256 f16  =   147,456 B
  _Float16* ow1p  = (_Float16*)(ws + 1542144); // 128*256 f16  =    65,536 B
  _Float16* ow2p  = (_Float16*)(ws + 1607680); // 256*128 f16  =    65,536 B

  const long accN = 2048L*128L;
  zero_kernel<<<(int)((accN + 255)/256), 256, 0, stream>>>(accum, accN);

  pack_kernel<<<192, 256, 0, stream>>>(mw1, mw1p, 177, 256, 6);
  pack_kernel<<<256, 256, 0, stream>>>(mw2, mw2p, 256, 256, 8);
  pack_kernel<<<128, 256, 0, stream>>>(mw3, mw3p, 256, 128, 8);
  pack_kernel<<<288, 256, 0, stream>>>(gw1, gw1p, 273, 256, 9);
  pack_kernel<<<128, 256, 0, stream>>>(ow1, ow1p, 128, 256, 4);
  pack_kernel<<<128, 256, 0, stream>>>(ow2, ow2p, 256, 128, 8);

  egate_kernel<<<1, 256, 0, stream>>>(e_feat, ew1, eb1, ew2, eb2, ew3, eb3, egate);

  // E = 262144 edges, 32 per wave, 1 wave per block -> 8192 blocks
  edge_kernel<<<8192, 32, 0, stream>>>(h, z, mask, att_src, att_dst, att_dist, ze,
                                       mw1p, mb1, mw2p, mb2, mw3p, mb3,
                                       gw1p, gb1, gw2, gb2, accum);

  // 2048*100 = 204800 rows, 32 per wave, 2 waves per block -> 3200 blocks
  out_kernel<<<3200, 64, 0, stream>>>(accum, egate, ow1p, ob1, ow2p, ob2, (float*)d_out);
}
